// AAEncoder_70196945486178
// MI455X (gfx1250) — matrix operand order repaired and checked by reading, not compiled, not run on hardware
//
#include <hip/hip_runtime.h>
#include <hip/hip_bf16.h>

// ---------------------------------------------------------------------------
// Problem constants (from reference)
// ---------------------------------------------------------------------------
#define N_NODES 640
#define T_STEPS 50
#define E_DIM   64
#define RADIUS2 2500.0f   // MAX_RADIUS^2

typedef __attribute__((ext_vector_type(16))) _Float16 v16h;
typedef __attribute__((ext_vector_type(8)))  float    v8f;
typedef _Float16 h16;

union FragU { v16h v; unsigned int u[8]; };
union H2U   { h16 h[2]; unsigned int u; };

// ---------------------------------------------------------------------------
// Dynamic LDS: single u32 view, addressed by 32-bit word indices only.
// (Direct indexing of the __shared__ array stays in LDS address space ->
//  ds_load/ds_store with 32-bit address math, no generic-pointer null checks.)
// ---------------------------------------------------------------------------
extern __shared__ __align__(16) unsigned int smemu[];

enum : int {
    STG_W   = 0,                      // 256x64 f32 staging
    QTILE_W = STG_W   + 256 * 64,     // 16x64 f32 q rows
    SCORE_W = QTILE_W + 16 * 64,      // 256x8 f32 scores
    B2_W    = SCORE_W + 256 * 8,      // na_b2
    BO_W    = B2_W    + 64,           // no_b
    BK_W    = BO_W    + 64,
    BV_W    = BK_W    + 64,
    W1X_W   = BV_W    + 64,           // na_w1 col0
    W1Y_W   = W1X_W   + 64,           // na_w1 col1
    B1_W    = W1Y_W   + 64,           // na_b1
    POSI_W  = B1_W    + 64,           // 16x2 f32
    POSJ_W  = POSI_W  + 32,
    PADI_W  = POSJ_W  + 32,           // 16 ints
    PADJ_W  = PADI_W  + 16,
    ADJ_W   = PADJ_W  + 16,           // 256 ints
    ACT_W   = ADJ_W   + 256,          // 256x64 f16 packed -> 8192 words
    WA_W    = ACT_W   + 8192,         // na_w2 [n][k] f16 packed
    WO_W    = WA_W    + 2048,         // no_w
    WK_W    = WO_W    + 2048,         // wk
    WV_W    = WK_W    + 2048,         // wv
    TOTAL_W = WV_W    + 2048
};

__device__ __forceinline__ float lds_f(int i)          { return __uint_as_float(smemu[i]); }
__device__ __forceinline__ void  lds_sf(int i, float v){ smemu[i] = __float_as_uint(v); }

// ---------------------------------------------------------------------------
// WMMA fragment loaders (CDNA5 16-bit layouts, wave32); word-indexed LDS.
// A: 16x32 (MxK). lane L: row = L%16; half = L/16.
//    VGPR 0..3 -> K = half*8 + 2v; VGPR 4..7 -> K = 16 + half*8 + 2(v-4).
//    Packed pairs => words [hsel*4 + v] and [8 + hsel*4 + v] (+ kstep*16).
// B: 32x16 (KxN), weight stored [n][k]. lane L: col = L%16.
//    Same K mapping as A: words [hsel*4 + v] and [8 + hsel*4 + v].
// ---------------------------------------------------------------------------
__device__ __forceinline__ v16h load_a_frag(int mbase, int lane, int kstep) {
    FragU fr;
    const int row_w = ACT_W + (mbase + (lane & 15)) * 32 + kstep * 16;
    const int h4    = ((lane >> 4) & 1) * 4;
#pragma unroll
    for (int v = 0; v < 4; ++v) fr.u[v] = smemu[row_w + h4 + v];
#pragma unroll
    for (int v = 0; v < 4; ++v) fr.u[4 + v] = smemu[row_w + 8 + h4 + v];
    return fr.v;
}

__device__ __forceinline__ v16h load_b_frag(int w_base, int nbase, int lane, int kstep) {
    FragU fr;
    const int col_w = w_base + (nbase + (lane & 15)) * 32 + kstep * 16 +
                      ((lane >> 4) & 1) * 4;
#pragma unroll
    for (int v = 0; v < 8; ++v) fr.u[v] = smemu[col_w + (v & 3) + ((v >> 2) << 3)];
    return fr.v;
}

// GEMM: (256x64 f16 act LDS) @ W^T (64x64 f16 LDS [n][k]) -> f32 stg (+bias[+hb]).
// Each wave handles M-tiles {2w, 2w+1}; 4 N-tiles; 2 K-steps of 32.
template <bool ADD_HB>
__device__ __forceinline__ void gemm_store(int w_base, int bias_w,
                                           const float* __restrict__ hb_g,
                                           int j0, int wave, int lane)
{
    v8f acc[2][4];
    const v8f vzero = {0.f, 0.f, 0.f, 0.f, 0.f, 0.f, 0.f, 0.f};
#pragma unroll
    for (int mt = 0; mt < 2; ++mt)
#pragma unroll
        for (int nt = 0; nt < 4; ++nt) acc[mt][nt] = vzero;

#pragma unroll
    for (int mt = 0; mt < 2; ++mt) {
        const int mb = (2 * wave + mt) * 16;
#pragma unroll
        for (int ks = 0; ks < 2; ++ks) {
            v16h a = load_a_frag(mb, lane, ks);
#pragma unroll
            for (int nt = 0; nt < 4; ++nt) {
                v16h b = load_b_frag(w_base, nt * 16, lane, ks);
                acc[mt][nt] = __builtin_amdgcn_wmma_f32_16x16x32_f16(
                    false, a, false, b, (short)0, acc[mt][nt], false, false);
            }
        }
    }

    // D layout: lane L -> n = L%16; element r -> m = r + 8*(L/16)
    const int n0 = lane & 15;
    const int mh = ((lane >> 4) & 1) * 8;
#pragma unroll
    for (int mt = 0; mt < 2; ++mt) {
        const int mb = (2 * wave + mt) * 16;
#pragma unroll
        for (int nt = 0; nt < 4; ++nt) {
            const int n = nt * 16 + n0;
            const float bn = lds_f(bias_w + n);
#pragma unroll
            for (int r = 0; r < 8; ++r) {
                const int p = mb + mh + r;          // pair index 0..255
                float val = acc[mt][nt][r] + bn;
                if (ADD_HB) val += hb_g[(j0 + (p & 15)) * 64 + n];
                lds_sf(STG_W + p * 64 + n, val);
            }
        }
    }
}

// Per-pair LayerNorm (row of 64): f32 stg -> packed f16 act, optional relu.
__device__ __forceinline__ void ln_rows_pack(int tid, bool do_relu) {
    const int row = STG_W + tid * 64;
    float s1 = 0.f, s2 = 0.f;
#pragma unroll
    for (int f = 0; f < 64; ++f) { float u = lds_f(row + f); s1 += u; s2 += u * u; }
    const float mean = s1 * (1.f / 64.f);
    const float var  = s2 * (1.f / 64.f) - mean * mean;
    const float rs   = rsqrtf(var + 1e-5f);
    const int arow = ACT_W + tid * 32;
#pragma unroll
    for (int f = 0; f < 64; f += 2) {
        float y0 = (lds_f(row + f)     - mean) * rs;
        float y1 = (lds_f(row + f + 1) - mean) * rs;
        if (do_relu) { y0 = fmaxf(y0, 0.f); y1 = fmaxf(y1, 0.f); }
        H2U pk; pk.h[0] = (h16)y0; pk.h[1] = (h16)y1;
        smemu[arow + (f >> 1)] = pk.u;
    }
}

// LayerNorm across a 64-entry static-LDS vector (phase A/C helper).
__device__ __forceinline__ float ln64(const float* v, int f) {
    float s1 = 0.f, s2 = 0.f;
    for (int k = 0; k < 64; ++k) { float u = v[k]; s1 += u; s2 += u * u; }
    const float m   = s1 * (1.f / 64.f);
    const float var = s2 * (1.f / 64.f) - m * m;
    return (v[f] - m) * rsqrtf(var + 1e-5f);
}

// ---------------------------------------------------------------------------
// Phase A: per-node precompute (center MLP, hn-projections, hb). 640 x 64.
// ---------------------------------------------------------------------------
__global__ void __launch_bounds__(64) node_pre(
    const float* __restrict__ positions, const unsigned char* __restrict__ bos_mask,
    const int* __restrict__ t_ptr,
    const float* __restrict__ ce_w1, const float* __restrict__ ce_b1,
    const float* __restrict__ ce_w2, const float* __restrict__ ce_b2,
    const float* __restrict__ ce_w3, const float* __restrict__ ce_b3,
    const float* __restrict__ nb_w1, const float* __restrict__ nb_b1,
    const float* __restrict__ nb_w2, const float* __restrict__ nb_b2,
    const float* __restrict__ wq, const float* __restrict__ bq,
    const float* __restrict__ wsm, const float* __restrict__ bsm,
    const float* __restrict__ whh, const float* __restrict__ bhh,
    const float* __restrict__ bos_token,
    float* __restrict__ ws_pos, int* __restrict__ ws_pad,
    float* __restrict__ ws_center, float* __restrict__ ws_hb,
    float* __restrict__ ws_q, float* __restrict__ ws_s, float* __restrict__ ws_ghh)
{
    const int i = blockIdx.x;
    const int f = threadIdx.x;
    const int t = *t_ptr;
    __shared__ float dp[2];
    __shared__ float va[64];
    __shared__ float vb[64];

    if (f < 2) {
        float pt = positions[(i * T_STEPS + t) * 2 + f];
        float pm = positions[(i * T_STEPS + t - 1) * 2 + f];
        dp[f] = pt - pm;
        ws_pos[i * 2 + f] = pt;
    }
    const int pad = bos_mask[i * T_STEPS + t] ? 1 : 0;
    if (f == 0) ws_pad[i] = pad;
    __syncthreads();

    float x = ce_b1[f] + dp[0] * ce_w1[f * 2] + dp[1] * ce_w1[f * 2 + 1];
    va[f] = x; __syncthreads();
    x = fmaxf(ln64(va, f), 0.f);
    __syncthreads(); va[f] = x; __syncthreads();

    float y = ce_b2[f];
    for (int k = 0; k < 64; ++k) y += va[k] * ce_w2[f * 64 + k];
    vb[f] = y; __syncthreads();
    y = fmaxf(ln64(vb, f), 0.f);
    __syncthreads(); vb[f] = y; __syncthreads();

    float z = ce_b3[f];
    for (int k = 0; k < 64; ++k) z += vb[k] * ce_w3[f * 64 + k];
    va[f] = z; __syncthreads();
    z = ln64(va, f);
    const float center = pad ? bos_token[t * 64 + f] : z;
    ws_center[i * 64 + f] = center;
    __syncthreads(); va[f] = center; __syncthreads();
    const float hn = ln64(va, f);
    __syncthreads(); va[f] = hn; __syncthreads();

    float q = bq[f], sv = bsm[f], g = bhh[f];
    for (int k = 0; k < 64; ++k) {
        const float h = va[k];
        q  += h * wq[f * 64 + k];
        sv += h * wsm[f * 64 + k];
        g  += h * whh[f * 64 + k];
    }
    ws_q[i * 64 + f]   = q;
    ws_s[i * 64 + f]   = sv;
    ws_ghh[i * 64 + f] = g;

    float u = nb_b1[f] + dp[0] * nb_w1[f * 2] + dp[1] * nb_w1[f * 2 + 1];
    __syncthreads(); va[f] = u; __syncthreads();
    u = fmaxf(ln64(va, f), 0.f);
    __syncthreads(); va[f] = u; __syncthreads();
    float hbv = nb_b2[f];
    for (int k = 0; k < 64; ++k) hbv += va[k] * nb_w2[f * 64 + k];
    ws_hb[i * 64 + f] = hbv;
}

// ---------------------------------------------------------------------------
// Phase B: fused pairwise MLP + k/v projection + masked online-softmax attn.
// grid = N/16 i-tiles, block = 256 threads (8 waves). Dynamic LDS (word view).
// ---------------------------------------------------------------------------
__global__ void __launch_bounds__(256) fused_nbr_attn(
    const float* __restrict__ pos,   // N x 2
    const int*   __restrict__ padv,  // N
    const float* __restrict__ qv,    // N x 64
    const float* __restrict__ hb,    // N x 64
    const float* __restrict__ na_w1, const float* __restrict__ na_b1,
    const float* __restrict__ na_w2, const float* __restrict__ na_b2,
    const float* __restrict__ no_w,  const float* __restrict__ no_b,
    const float* __restrict__ wk,    const float* __restrict__ bk,
    const float* __restrict__ wv,    const float* __restrict__ bv,
    float* __restrict__ agg_out)     // N x 64
{
    const int tid  = threadIdx.x;
    const int lane = tid & 31;
    const int wave = tid >> 5;
    const int i0   = blockIdx.x * 16;

    // Preload weights as packed f16 (two consecutive k per word, [n][k] rows).
    for (int idx = tid; idx < 2048; idx += 256) {
        H2U a, o, k2, v2;
        a.h[0]  = (h16)na_w2[2 * idx]; a.h[1]  = (h16)na_w2[2 * idx + 1];
        o.h[0]  = (h16)no_w[2 * idx];  o.h[1]  = (h16)no_w[2 * idx + 1];
        k2.h[0] = (h16)wk[2 * idx];    k2.h[1] = (h16)wk[2 * idx + 1];
        v2.h[0] = (h16)wv[2 * idx];    v2.h[1] = (h16)wv[2 * idx + 1];
        smemu[WA_W + idx] = a.u;
        smemu[WO_W + idx] = o.u;
        smemu[WK_W + idx] = k2.u;
        smemu[WV_W + idx] = v2.u;
    }
    for (int idx = tid; idx < 1024; idx += 256)
        lds_sf(QTILE_W + idx, qv[i0 * 64 + idx]);
    if (tid < 64) {
        lds_sf(B2_W + tid, na_b2[tid]);
        lds_sf(BO_W + tid, no_b[tid]);
        lds_sf(BK_W + tid, bk[tid]);
        lds_sf(BV_W + tid, bv[tid]);
        lds_sf(W1X_W + tid, na_w1[tid * 2 + 0]);
        lds_sf(W1Y_W + tid, na_w1[tid * 2 + 1]);
        lds_sf(B1_W + tid, na_b1[tid]);
    }
    if (tid < 16) {
        lds_sf(POSI_W + tid * 2 + 0, pos[(i0 + tid) * 2 + 0]);
        lds_sf(POSI_W + tid * 2 + 1, pos[(i0 + tid) * 2 + 1]);
        smemu[PADI_W + tid] = (unsigned int)padv[i0 + tid];
    }
    __syncthreads();

    // Online-softmax state for owner threads (tid<128): (row, head)
    const int own_i = tid >> 3;
    const int own_h = tid & 7;
    float mrun = -3.0e38f, srun = 0.0f, vacc[8];
    unsigned int hasany = 0;
#pragma unroll
    for (int d = 0; d < 8; ++d) vacc[d] = 0.0f;

    for (int j0 = 0; j0 < N_NODES; j0 += 16) {
        if (tid < 16) {
            lds_sf(POSJ_W + tid * 2 + 0, pos[(j0 + tid) * 2 + 0]);
            lds_sf(POSJ_W + tid * 2 + 1, pos[(j0 + tid) * 2 + 1]);
            smemu[PADJ_W + tid] = (unsigned int)padv[j0 + tid];
        }
        __syncthreads();

        // ---- u1 = relu(ln(rel @ na_w1^T + na_b1)), adjacency ----
        {
            const int p  = tid;
            const int il = p >> 4, jl = p & 15;
            const float rx = lds_f(POSJ_W + jl * 2 + 0) - lds_f(POSI_W + il * 2 + 0);
            const float ry = lds_f(POSJ_W + jl * 2 + 1) - lds_f(POSI_W + il * 2 + 1);
            const float d2 = rx * rx + ry * ry;
            const unsigned int a_ok =
                (d2 <= RADIUS2) && (smemu[PADI_W + il] == 0u) &&
                (smemu[PADJ_W + jl] == 0u) && ((i0 + il) != (j0 + jl));
            smemu[ADJ_W + p] = a_ok;

            const int row = STG_W + p * 64;
            float s1 = 0.f, s2 = 0.f;
#pragma unroll
            for (int f = 0; f < 64; ++f) {
                const float u = rx * lds_f(W1X_W + f) + ry * lds_f(W1Y_W + f) +
                                lds_f(B1_W + f);
                lds_sf(row + f, u); s1 += u; s2 += u * u;
            }
            const float mean = s1 * (1.f / 64.f);
            const float var  = s2 * (1.f / 64.f) - mean * mean;
            const float rs   = rsqrtf(var + 1e-5f);
            const int arow = ACT_W + p * 32;
#pragma unroll
            for (int f = 0; f < 64; f += 2) {
                const float y0 = fmaxf((lds_f(row + f)     - mean) * rs, 0.f);
                const float y1 = fmaxf((lds_f(row + f + 1) - mean) * rs, 0.f);
                H2U pk; pk.h[0] = (h16)y0; pk.h[1] = (h16)y1;
                smemu[arow + (f >> 1)] = pk.u;
            }
        }
        __syncthreads();

        // ---- GEMM1: u1 @ na_w2^T + na_b2 + hb[j]  ->  ln,relu ----
        gemm_store<true>(WA_W, B2_W, hb, j0, wave, lane);
        __syncthreads();
        ln_rows_pack(tid, true);
        __syncthreads();

        // ---- GEMM2: (.) @ no_w^T + no_b  ->  ln  => nbr ----
        gemm_store<false>(WO_W, BO_W, nullptr, j0, wave, lane);
        __syncthreads();
        ln_rows_pack(tid, false);
        __syncthreads();

        // ---- GEMM3: k = nbr @ wk^T + bk ----
        gemm_store<false>(WK_W, BK_W, nullptr, j0, wave, lane);
        __syncthreads();

        // ---- scores (per pair, per head) ----
        {
            const int p  = tid;
            const int il = p >> 4;
            const float scale = 0.35355339059327373f;  // 1/sqrt(8)
            const unsigned int a_ok = smemu[ADJ_W + p];
#pragma unroll
            for (int h = 0; h < 8; ++h) {
                float s = 0.f;
#pragma unroll
                for (int d = 0; d < 8; ++d)
                    s += lds_f(QTILE_W + il * 64 + h * 8 + d) *
                         lds_f(STG_W + p * 64 + h * 8 + d);
                lds_sf(SCORE_W + p * 8 + h, a_ok ? (s * scale) : -1.0e9f);
            }
        }
        __syncthreads();

        // ---- GEMM4: v = nbr @ wv^T + bv ----
        gemm_store<false>(WV_W, BV_W, nullptr, j0, wave, lane);
        __syncthreads();

        // ---- online softmax + weighted V accumulation ----
        if (tid < 128) {
            for (int jj = 0; jj < 16; ++jj) {
                const int p = own_i * 16 + jj;
                hasany |= smemu[ADJ_W + p];
                const float sc = lds_f(SCORE_W + p * 8 + own_h);
                const float nm = fmaxf(mrun, sc);
                const float c  = __expf(mrun - nm);
                const float e  = __expf(sc - nm);
                srun = srun * c + e;
#pragma unroll
                for (int d = 0; d < 8; ++d)
                    vacc[d] = vacc[d] * c + e * lds_f(STG_W + p * 64 + own_h * 8 + d);
                mrun = nm;
            }
        }
        __syncthreads();
    }

    if (tid < 128) {
        const float inv = hasany ? (1.0f / srun) : 0.0f;
#pragma unroll
        for (int d = 0; d < 8; ++d)
            agg_out[(i0 + own_i) * 64 + own_h * 8 + d] = vacc[d] * inv;
    }
}

// ---------------------------------------------------------------------------
// Phase C: gate, output projection, MLP residual. 640 x 64.
// ---------------------------------------------------------------------------
__global__ void __launch_bounds__(64) node_post(
    const float* __restrict__ ws_center, const float* __restrict__ ws_agg,
    const float* __restrict__ ws_s, const float* __restrict__ ws_ghh,
    const float* __restrict__ w_ih, const float* __restrict__ b_ih,
    const float* __restrict__ wo, const float* __restrict__ bo,
    const float* __restrict__ m_w1, const float* __restrict__ m_b1,
    const float* __restrict__ m_w2, const float* __restrict__ m_b2,
    float* __restrict__ out)
{
    const int i = blockIdx.x;
    const int f = threadIdx.x;
    __shared__ float aggs[64];
    __shared__ float attn[64];
    __shared__ float xv[64];
    __shared__ float h2v[64];
    __shared__ float hid[256];

    aggs[f] = ws_agg[i * 64 + f];
    __syncthreads();

    float gin = b_ih[f] + ws_ghh[i * 64 + f];
    for (int k = 0; k < 64; ++k) gin += aggs[k] * w_ih[f * 64 + k];
    const float gate = 1.0f / (1.0f + __expf(-gin));
    const float a = aggs[f];
    attn[f] = a + gate * (ws_s[i * 64 + f] - a);
    __syncthreads();

    float x = ws_center[i * 64 + f] + bo[f];
    for (int k = 0; k < 64; ++k) x += attn[k] * wo[f * 64 + k];
    xv[f] = x; __syncthreads();
    const float h2 = ln64(xv, f);
    __syncthreads();
    h2v[f] = h2; __syncthreads();

    for (int r = 0; r < 4; ++r) {
        const int of = r * 64 + f;
        float hh = m_b1[of];
        for (int k = 0; k < 64; ++k) hh += h2v[k] * m_w1[of * 64 + k];
        hid[of] = fmaxf(hh, 0.f);
    }
    __syncthreads();

    float o = x + m_b2[f];
    for (int k = 0; k < 256; ++k) o += hid[k] * m_w2[f * 256 + k];
    out[i * 64 + f] = o;
}

// ---------------------------------------------------------------------------
// Host launcher
// ---------------------------------------------------------------------------
extern "C" void kernel_launch(void* const* d_in, const int* in_sizes, int n_in,
                              void* d_out, int out_size, void* d_ws, size_t ws_size,
                              hipStream_t stream)
{
    (void)in_sizes; (void)n_in; (void)out_size; (void)ws_size;

    const float*         positions = (const float*)d_in[0];
    const unsigned char* bos_mask  = (const unsigned char*)d_in[1];
    const int*           t_ptr     = (const int*)d_in[2];
    const float* ce_w1 = (const float*)d_in[3];  const float* ce_b1 = (const float*)d_in[4];
    const float* ce_w2 = (const float*)d_in[5];  const float* ce_b2 = (const float*)d_in[6];
    const float* ce_w3 = (const float*)d_in[7];  const float* ce_b3 = (const float*)d_in[8];
    const float* na_w1 = (const float*)d_in[9];  const float* na_b1 = (const float*)d_in[10];
    const float* na_w2 = (const float*)d_in[11]; const float* na_b2 = (const float*)d_in[12];
    const float* nb_w1 = (const float*)d_in[13]; const float* nb_b1 = (const float*)d_in[14];
    const float* nb_w2 = (const float*)d_in[15]; const float* nb_b2 = (const float*)d_in[16];
    const float* no_w  = (const float*)d_in[17]; const float* no_b  = (const float*)d_in[18];
    const float* wq    = (const float*)d_in[19]; const float* bq    = (const float*)d_in[20];
    const float* wk    = (const float*)d_in[21]; const float* bk    = (const float*)d_in[22];
    const float* wv    = (const float*)d_in[23]; const float* bv    = (const float*)d_in[24];
    const float* wsm   = (const float*)d_in[25]; const float* bsm   = (const float*)d_in[26];
    const float* w_ih  = (const float*)d_in[27]; const float* b_ih  = (const float*)d_in[28];
    const float* w_hh  = (const float*)d_in[29]; const float* b_hh  = (const float*)d_in[30];
    const float* wo    = (const float*)d_in[31]; const float* bo    = (const float*)d_in[32];
    const float* m_w1  = (const float*)d_in[33]; const float* m_b1  = (const float*)d_in[34];
    const float* m_w2  = (const float*)d_in[35]; const float* m_b2  = (const float*)d_in[36];
    const float* bos_token = (const float*)d_in[37];

    // Workspace partition (floats)
    float* wsf = (float*)d_ws;
    size_t off = 0;
    float* ws_pos    = wsf + off;            off += N_NODES * 2;
    int*   ws_pad    = (int*)(wsf + off);    off += N_NODES;
    float* ws_center = wsf + off;            off += N_NODES * E_DIM;
    float* ws_hb     = wsf + off;            off += N_NODES * E_DIM;
    float* ws_q      = wsf + off;            off += N_NODES * E_DIM;
    float* ws_s      = wsf + off;            off += N_NODES * E_DIM;
    float* ws_ghh    = wsf + off;            off += N_NODES * E_DIM;
    float* ws_agg    = wsf + off;            off += N_NODES * E_DIM;

    node_pre<<<N_NODES, 64, 0, stream>>>(
        positions, bos_mask, t_ptr,
        ce_w1, ce_b1, ce_w2, ce_b2, ce_w3, ce_b3,
        nb_w1, nb_b1, nb_w2, nb_b2,
        wq, bq, wsm, bsm, w_hh, b_hh, bos_token,
        ws_pos, ws_pad, ws_center, ws_hb, ws_q, ws_s, ws_ghh);

    const size_t smem = (size_t)TOTAL_W * 4;   // 146560 bytes
    (void)hipFuncSetAttribute((const void*)fused_nbr_attn,
                              hipFuncAttributeMaxDynamicSharedMemorySize,
                              (int)smem);
    fused_nbr_attn<<<N_NODES / 16, 256, smem, stream>>>(
        ws_pos, ws_pad, ws_q, ws_hb,
        na_w1, na_b1, na_w2, na_b2, no_w, no_b,
        wk, bk, wv, bv, ws_agg);

    node_post<<<N_NODES, 64, 0, stream>>>(
        ws_center, ws_agg, ws_s, ws_ghh,
        w_ih, b_ih, wo, bo, m_w1, m_b1, m_w2, m_b2,
        (float*)d_out);
}
